// bioRNN_Model_82600811037106
// MI455X (gfx1250) — matrix operand order repaired and checked by reading, not compiled, not run on hardware
//
#include <hip/hip_runtime.h>

// ---- problem constants (match reference) ----
#define TT    400
#define BB    256
#define NINN  64
#define HH    512
#define HP    (HH + 4)    // padded fp32 h-state row (516 mod 64 = 4 -> no bank conflicts)
#define NOUTT 16
#define BS    16          // batch rows per workgroup
#define KTOT  (NINN + HH) // 576 = concatenated K
#define NKT   (KTOT / 32) // 18 k-tiles for the main GEMM
#define ROWD  292         // dwords per A row in LDS (288 data + 4 pad)
#define NEXC  409         // int(0.8*512)

typedef __attribute__((ext_vector_type(16))) __bf16       v16bf;
typedef __attribute__((ext_vector_type(8)))  float        v8f;
typedef __attribute__((ext_vector_type(4)))  float        v4f;
typedef __attribute__((ext_vector_type(4)))  unsigned int v4u;
typedef __attribute__((ext_vector_type(8)))  unsigned int v8u;

__device__ __forceinline__ unsigned int pack2_bf16(float a, float b) {
  unsigned int ua = __float_as_uint(a);
  unsigned int ub = __float_as_uint(b);
  ua = (ua + 0x7FFFu + ((ua >> 16) & 1u)) >> 16;   // RNE
  ub = (ub + 0x7FFFu + ((ub >> 16) & 1u)) >> 16;
  return (ub << 16) | (ua & 0xFFFFu);              // first element in [15:0]
}

__device__ __forceinline__ v16bf load_frag(const unsigned int* lo,
                                           const unsigned int* hi) {
  v4u a = *(const v4u*)lo;
  v4u b = *(const v4u*)hi;
  v8u w;
  w[0] = a[0]; w[1] = a[1]; w[2] = a[2]; w[3] = a[3];
  w[4] = b[0]; w[5] = b[1]; w[6] = b[2]; w[7] = b[3];
  return __builtin_bit_cast(v16bf, w);
}

// Build an A-fragment (16x32 bf16, wave32 layout) straight from the fp32
// h-state row: dwords 0..3 hold K = kbase+khalf*8+{0..7}, dwords 4..7 hold
// K = kbase+16+khalf*8+{0..7} (bf16 pairs).
__device__ __forceinline__ v16bf frag_from_h(const float* rowp, int kbase,
                                             int khalf) {
  const float* p0 = rowp + kbase + khalf * 8;
  v4f a0 = *(const v4f*)p0;
  v4f a1 = *(const v4f*)(p0 + 4);
  const float* p1 = p0 + 16;
  v4f b0 = *(const v4f*)p1;
  v4f b1 = *(const v4f*)(p1 + 4);
  v8u w;
  w[0] = pack2_bf16(a0[0], a0[1]); w[1] = pack2_bf16(a0[2], a0[3]);
  w[2] = pack2_bf16(a1[0], a1[1]); w[3] = pack2_bf16(a1[2], a1[3]);
  w[4] = pack2_bf16(b0[0], b0[1]); w[5] = pack2_bf16(b0[2], b0[3]);
  w[6] = pack2_bf16(b1[0], b1[1]); w[7] = pack2_bf16(b1[2], b1[3]);
  return __builtin_bit_cast(v16bf, w);
}

// ---------------- weight packing prologue ----------------
// Wcat = [relu(W_in); relu(W_rnn)*ei_sign], bf16, packed directly into the
// wave32 WMMA B-fragment layout: block(ntile,ktile) of 256 dwords, dword
// index = lane*8 + i, lane -> (N = lane&15, kgrp = lane>>4), dword i holds
// bf16 pair (K = ktile*32 + kgrp*8 + (i&3)*2 + (i>>2)*16, K+1).
__device__ __forceinline__ float eff_w(const float* Win, const float* Wrnn,
                                       int k, int n) {
  if (k < NINN) {
    float v = Win[k * HH + n];
    return v > 0.f ? v : 0.f;
  }
  int kr = k - NINN;
  float v = Wrnn[kr * HH + n];
  v = v > 0.f ? v : 0.f;
  return (kr < NEXC) ? v : -v;
}

__global__ void pack_wcat(const float* __restrict__ Win,
                          const float* __restrict__ Wrnn,
                          unsigned int* __restrict__ wcat) {
  int g     = blockIdx.x * 256 + threadIdx.x;   // 576 blocks -> 147456 dwords
  int block = g >> 8;                           // ntile*18 + ktile
  int li    = g & 255;                          // lane*8 + i
  int l     = li >> 3, i = li & 7;
  int ntile = block / NKT, kt = block % NKT;
  int n  = ntile * 16 + (l & 15);
  int k0 = kt * 32 + (l >> 4) * 8 + (i & 3) * 2 + (i >> 2) * 16;
  wcat[g] = pack2_bf16(eff_w(Win, Wrnn, k0, n), eff_w(Win, Wrnn, k0 + 1, n));
}

__global__ void pack_wout(const float* __restrict__ Wout,
                          unsigned int* __restrict__ wpk) {
  int g  = blockIdx.x * 256 + threadIdx.x;      // 16 blocks -> 4096 dwords
  int kt = g >> 8;
  int li = g & 255;
  int l  = li >> 3, i = li & 7;
  int n  = l & 15;
  int k0 = kt * 32 + (l >> 4) * 8 + (i & 3) * 2 + (i >> 2) * 16;
  float a = Wout[k0 * NOUTT + n];       a = a > 0.f ? a : 0.f;
  float b = Wout[(k0 + 1) * NOUTT + n]; b = b > 0.f ? b : 0.f;
  wpk[g] = pack2_bf16(a, b);
}

// ---------------- main persistent recurrent kernel ----------------
__global__ void __launch_bounds__(256)
bio_rnn_main(const float* __restrict__ x, const float* __restrict__ noise,
             const float* __restrict__ brnn_g, const float* __restrict__ bout_g,
             const unsigned int* __restrict__ wcat,
             const unsigned int* __restrict__ wout,
             float* __restrict__ out_logits, float* __restrict__ out_h,
             float* __restrict__ out_sx, float* __restrict__ out_su) {
  __shared__ float        sh_h[BS][HP];                       // 33 KB (padded)
  __shared__ __align__(16) unsigned int sh_A[BS][ROWD];       // ~18.7 KB (bf16 pairs)
  __shared__ float        sh_part[8][256];                    // 8 KB logits partials
  __shared__ float        sh_brnn[HH];
  __shared__ float        sh_bout[NOUTT];

  const int tid   = threadIdx.x;
  const int wave  = tid >> 5;     // 8 waves (wave32)
  const int lane  = tid & 31;
  const int khalf = lane >> 4;
  const int nlo   = lane & 15;
  const int bbase = blockIdx.x * BS;

  // per-thread synapse state: columns j0=2*tid (facilitating), j1=2*tid+1
  // (depressing), across all BS rows — lives entirely in VGPRs.
  const int   j0 = 2 * tid, j1 = 2 * tid + 1;
  const float UE = 0.15f, UO = 0.45f;
  const float ASTF_E = 10.f / 1500.f, ASTD_E = 10.f / 200.f;
  const float ASTF_O = 10.f / 200.f,  ASTD_O = 10.f / 1500.f;
  const float DTS = 0.01f, AL = 0.1f, OMAL = 0.9f;

  float rsx[2 * BS], rsu[2 * BS];
#pragma unroll
  for (int r = 0; r < BS; ++r) {
    rsx[2 * r] = 1.f;  rsx[2 * r + 1] = 1.f;
    rsu[2 * r] = UE;   rsu[2 * r + 1] = UO;
  }
  for (int idx = tid; idx < BS * HH; idx += 256)
    sh_h[idx >> 9][idx & (HH - 1)] = 0.1f;
  for (int idx = tid; idx < HH; idx += 256) sh_brnn[idx] = brnn_g[idx];
  if (tid < NOUTT) sh_bout[tid] = bout_g[tid];
  __syncthreads();

  for (int t = 0; t < TT; ++t) {
    // ---- phase E: STP dynamics, stage A = [x_t | h_post] in bf16 ----
    const float* nzb = noise + ((size_t)t * BB + bbase) * HH;
    __builtin_prefetch(nzb + tid * 32, 0, 0);   // warm this step's noise tile
    {
      float* sxp = out_sx + ((size_t)t * BB + bbase) * HH;
      float* sup = out_su + ((size_t)t * BB + bbase) * HH;
#pragma unroll
      for (int r = 0; r < BS; ++r) {
        float h0 = sh_h[r][j0], h1 = sh_h[r][j1];
        float sx0 = rsx[2 * r], su0 = rsu[2 * r];
        float sx1 = rsx[2 * r + 1], su1 = rsu[2 * r + 1];
        float nsx0 = sx0 + ASTD_E * (1.f - sx0) - DTS * su0 * sx0 * h0;
        float nsu0 = su0 + ASTF_E * (UE - su0) + DTS * UE * (1.f - su0) * h0;
        float nsx1 = sx1 + ASTD_O * (1.f - sx1) - DTS * su1 * sx1 * h1;
        float nsu1 = su1 + ASTF_O * (UO - su1) + DTS * UO * (1.f - su1) * h1;
        nsx0 = fminf(fmaxf(nsx0, 0.f), 1.f); nsu0 = fminf(fmaxf(nsu0, 0.f), 1.f);
        nsx1 = fminf(fmaxf(nsx1, 0.f), 1.f); nsu1 = fminf(fmaxf(nsu1, 0.f), 1.f);
        rsx[2 * r] = nsx0; rsu[2 * r] = nsu0;
        rsx[2 * r + 1] = nsx1; rsu[2 * r + 1] = nsu1;
        float hp0 = nsu0 * nsx0 * h0, hp1 = nsu1 * nsx1 * h1;
        sh_A[r][32 + tid] = pack2_bf16(hp0, hp1);   // K cols 64..575
        size_t o = (size_t)r * HH + (size_t)j0;
        sxp[o] = nsx0; sxp[o + 1] = nsx1;
        sup[o] = nsu0; sup[o + 1] = nsu1;
      }
      const float* xt = x + ((size_t)t * BB + bbase) * NINN;
#pragma unroll
      for (int k = 0; k < 2; ++k) {                 // 512 dwords of x_t
        int d = tid + k * 256;
        int r = d >> 5, c = d & 31;
        sh_A[r][c] = pack2_bf16(xt[r * NINN + 2 * c], xt[r * NINN + 2 * c + 1]);
      }
    }
    __syncthreads();

    // ---- phase G: acc[16x512] = [x|h_post] @ Wcat (bf16 WMMA, f32 acc) ----
    v8f acc0 = {0,0,0,0,0,0,0,0}, acc1 = {0,0,0,0,0,0,0,0};
    v8f acc2 = {0,0,0,0,0,0,0,0}, acc3 = {0,0,0,0,0,0,0,0};
    {
      const unsigned int* arow = &sh_A[nlo][0];
      const unsigned int* wbase = wcat + (size_t)(wave * 4) * NKT * 256;
      for (int kt = 0; kt < NKT; ++kt) {
        const unsigned int* ap = arow + kt * 16 + khalf * 4;
        v16bf af = load_frag(ap, ap + 8);
        const unsigned int* bp = wbase + kt * 256 + lane * 8;
        v16bf b0 = load_frag(bp, bp + 4);
        acc0 = __builtin_amdgcn_wmma_f32_16x16x32_bf16(false, af, false, b0,
                                                       (short)0, acc0, false, false);
        const unsigned int* bp1 = bp + NKT * 256;
        v16bf b1 = load_frag(bp1, bp1 + 4);
        acc1 = __builtin_amdgcn_wmma_f32_16x16x32_bf16(false, af, false, b1,
                                                       (short)0, acc1, false, false);
        const unsigned int* bp2 = bp + 2 * NKT * 256;
        v16bf b2 = load_frag(bp2, bp2 + 4);
        acc2 = __builtin_amdgcn_wmma_f32_16x16x32_bf16(false, af, false, b2,
                                                       (short)0, acc2, false, false);
        const unsigned int* bp3 = bp + 3 * NKT * 256;
        v16bf b3 = load_frag(bp3, bp3 + 4);
        acc3 = __builtin_amdgcn_wmma_f32_16x16x32_bf16(false, af, false, b3,
                                                       (short)0, acc3, false, false);
      }
    }

    // ---- phase H: leaky integration + relu, write h state + h_seq ----
    {
      float* hseq = out_h + ((size_t)t * BB + bbase) * HH;
      auto epi = [&](v8f a, int j) {
        int N = wave * 64 + j * 16 + nlo;
#pragma unroll
        for (int v = 0; v < 8; ++v) {
          int M = v + 8 * khalf;
          float pre = a[v] + sh_brnn[N];
          float hn  = OMAL * sh_h[M][N] + AL * pre + nzb[(size_t)M * HH + N];
          hn = fmaxf(hn, 0.f);
          sh_h[M][N] = hn;
          hseq[(size_t)M * HH + N] = hn;
        }
      };
      epi(acc0, 0); epi(acc1, 1); epi(acc2, 2); epi(acc3, 3);
    }
    __syncthreads();

    // ---- phase O: logits = h_new @ Wout, K split across the 8 waves ----
    // A-fragments are built directly from the fp32 h state (padded rows ->
    // conflict-free ds_load_b128), partial 16x16 tiles reduced via LDS.
    {
      v8f lacc = {0,0,0,0,0,0,0,0};
      const float* hrow = &sh_h[nlo][0];
#pragma unroll
      for (int kk = 0; kk < 2; ++kk) {
        int kt = wave * 2 + kk;
        v16bf af = frag_from_h(hrow, kt * 32, khalf);
        const unsigned int* bp = wout + kt * 256 + lane * 8;
        v16bf bf = load_frag(bp, bp + 4);
        lacc = __builtin_amdgcn_wmma_f32_16x16x32_bf16(false, af, false, bf,
                                                       (short)0, lacc, false, false);
      }
#pragma unroll
      for (int v = 0; v < 8; ++v) sh_part[wave][v * 32 + lane] = lacc[v];
    }
    __syncthreads();
    {
      float s = sh_part[0][tid] + sh_part[1][tid] + sh_part[2][tid] +
                sh_part[3][tid] + sh_part[4][tid] + sh_part[5][tid] +
                sh_part[6][tid] + sh_part[7][tid];
      int v = tid >> 5, l = tid & 31;
      int M = v + 8 * ((l >> 4) & 1);
      int N = l & 15;
      out_logits[((size_t)t * BB + bbase + M) * NOUTT + N] = s + sh_bout[N];
    }
  }
}

extern "C" void kernel_launch(void* const* d_in, const int* in_sizes, int n_in,
                              void* d_out, int out_size, void* d_ws, size_t ws_size,
                              hipStream_t stream) {
  const float* x     = (const float*)d_in[0];
  const float* noise = (const float*)d_in[1];
  const float* Win   = (const float*)d_in[2];
  const float* Wrnn  = (const float*)d_in[3];
  const float* brnn  = (const float*)d_in[4];
  const float* Wout  = (const float*)d_in[5];
  const float* bout  = (const float*)d_in[6];

  unsigned int* wcat = (unsigned int*)d_ws;   // 147456 dwords (576 KB)
  unsigned int* wpk  = wcat + 147456;         // 4096 dwords  (16 KB)

  float* o          = (float*)d_out;          // (logits, h, sx, su) flat
  float* out_logits = o;
  float* out_h      = o + (size_t)TT * BB * NOUTT;
  float* out_sx     = out_h + (size_t)TT * BB * HH;
  float* out_su     = out_sx + (size_t)TT * BB * HH;

  pack_wcat<<<(32 * NKT), 256, 0, stream>>>(Win, Wrnn, wcat);
  pack_wout<<<16, 256, 0, stream>>>(Wout, wpk);
  bio_rnn_main<<<BB / BS, 256, 0, stream>>>(x, noise, brnn, bout, wcat, wpk,
                                            out_logits, out_h, out_sx, out_su);
}
